// EGRUDDVSModel_72249939853964
// MI455X (gfx1250) — compile-verified, matrix operand order checked by reading, not run on hardware
//
#include <hip/hip_runtime.h>
#include <hip/hip_bf16.h>

typedef __attribute__((ext_vector_type(8)))  _Float16 v8h;
typedef __attribute__((ext_vector_type(16))) _Float16 v16h;
typedef __attribute__((ext_vector_type(8)))  float    v8f;

#define BB 16
#define TT 128
#define HH 1024
#define G3 3072
#define MT 2048   // B*T

// recurrence geometry: 64 slices of H (16 cols each), 2 waves per WG -> 32 WGs
#define RWAVES 2
#define RWGS   32
// per-wave LDS for Whh slice: 3 gates * 32 ksteps * 1KB = 96KB; WG = 192KB
#define LDS_PER_GATE  (32 * 1024)
#define LDS_PER_WAVE  (3 * LDS_PER_GATE)
#define LDS_TOTAL     (RWAVES * LDS_PER_WAVE)

__device__ inline v8f zero8() {
  v8f z = {0.f,0.f,0.f,0.f,0.f,0.f,0.f,0.f};
  return z;
}

union V16U { v16h v; v8h h[2]; };

// Per-lane share of a 16x32 f16 A-fragment (or 32x16 B-fragment, same packing
// indexed by column) from a row-major row pointer advanced to the K-base.
// Lanes 0-15 own K {0..7,16..23}; lanes 16-31 own K {8..15,24..31}.
__device__ inline v16h load_frag(const _Float16* __restrict__ p, int hi) {
  V16U u;
  u.h[0] = *(const v8h*)(p + hi * 8);
  u.h[1] = *(const v8h*)(p + 16 + hi * 8);
  return u.v;
}

// B-fragment from LDS, pre-permuted at fill time:
// layout per gate: [kstep][chunk(2)][lane(32)] of 16B  ->  two ds_load_b128.
__device__ inline v16h lds_frag(const char* gbase, int ks, int lane) {
  V16U u;
  u.h[0] = *(const v8h*)(gbase + ks * 1024 + lane * 16);
  u.h[1] = *(const v8h*)(gbase + ks * 1024 + 512 + lane * 16);
  return u.v;
}

__device__ inline v8f wmma_f16(v16h a, v16h b, v8f c) {
  return __builtin_amdgcn_wmma_f32_16x16x32_f16(false, a, false, b, (short)0, c, false, false);
}

// ---------------------------------------------------------------------------
// 1) Preprocess: [B,T,2,128,128] -> maxpool4x4 -> mean over polarity -> f16
// ---------------------------------------------------------------------------
__global__ void preprocess_kernel(const float* __restrict__ in, _Float16* __restrict__ xf) {
  int id = blockIdx.x * blockDim.x + threadIdx.x;   // < 2048*1024
  int r = id >> 10;          // b*T + t
  int c = id & 1023;
  int fy = c >> 5, fx = c & 31;
  float acc = 0.f;
#pragma unroll
  for (int p = 0; p < 2; ++p) {
    const float* base = in + ((size_t)(r * 2 + p) * 128 + (size_t)fy * 4) * 128 + fx * 4;
    float mx = -3.402823466e38f;
#pragma unroll
    for (int ry = 0; ry < 4; ++ry) {
      float4 v = *(const float4*)(base + ry * 128);
      mx = fmaxf(mx, fmaxf(fmaxf(v.x, v.y), fmaxf(v.z, v.w)));
    }
    acc += mx;
  }
  xf[id] = (_Float16)(0.5f * acc);
}

// ---------------------------------------------------------------------------
// 2) f32 -> f16 weight conversion
// ---------------------------------------------------------------------------
__global__ void cvt_f16_kernel(const float* __restrict__ src, _Float16* __restrict__ dst, int n) {
  int i = blockIdx.x * blockDim.x + threadIdx.x;
  if (i < n) dst[i] = (_Float16)src[i];
}

// ---------------------------------------------------------------------------
// 3) Input-side GEMM: Y[2048,3072] = X[2048,1024](f16) * W[3072,1024]^T + bias
//    One wave per 16x64 output strip. K unrolled by 2; a sched_barrier pins
//    the grouped 20-load clause above the 8-WMMA chain so waits are partial.
//    (GEMM latency is additionally hidden by ~6 resident blocks per WGP.)
// ---------------------------------------------------------------------------
__global__ void __launch_bounds__(256)
__attribute__((amdgpu_waves_per_eu(2)))
gemm_xg_kernel(const _Float16* __restrict__ X,
               const _Float16* __restrict__ W,
               const float* __restrict__ bias,
               float* __restrict__ Y) {
  const int lane = threadIdx.x & 31;
  const int wid  = (blockIdx.x * blockDim.x + threadIdx.x) >> 5;  // 0..6143
  const int NSTRIP = G3 / 64;                                     // 48
  const int mtile  = wid / NSTRIP;                                // block's waves share mtile
  const int nstrip = wid % NSTRIP;
  const int hi = lane >> 4;
  const int ln = lane & 15;

  const _Float16* arow  = X + (size_t)(mtile * 16 + ln) * HH;
  const _Float16* wbase = W + (size_t)(nstrip * 64 + ln) * HH;
  v8f acc[4];
#pragma unroll
  for (int s = 0; s < 4; ++s) acc[s] = zero8();

  for (int k0 = 0; k0 < HH; k0 += 64) {
    v16h a0 = load_frag(arow + k0, hi);
    v16h a1 = load_frag(arow + k0 + 32, hi);
    v16h b0[4], b1[4];
#pragma unroll
    for (int s = 0; s < 4; ++s) {
      b0[s] = load_frag(wbase + (size_t)s * 16 * HH + k0, hi);
      b1[s] = load_frag(wbase + (size_t)s * 16 * HH + k0 + 32, hi);
    }
    __builtin_amdgcn_sched_barrier(0);   // loads stay above, WMMAs below
#pragma unroll
    for (int s = 0; s < 4; ++s) acc[s] = wmma_f16(a0, b0[s], acc[s]);
#pragma unroll
    for (int s = 0; s < 4; ++s) acc[s] = wmma_f16(a1, b1[s], acc[s]);
  }
#pragma unroll
  for (int s = 0; s < 4; ++s) {
    int col = nstrip * 64 + s * 16 + ln;
    float bv = bias[col];
#pragma unroll
    for (int v = 0; v < 8; ++v) {
      int row = mtile * 16 + v + 8 * hi;
      Y[(size_t)row * G3 + col] = acc[s][v] + bv;
    }
  }
}

// ---------------------------------------------------------------------------
// 4) Persistent GRU recurrence. 32 WGs x 64 threads = 64 waves; each wave
//    owns a 16-wide slice of H. Whh slice (96KB/wave) staged ONCE into LDS in
//    fragment order. The K-loop is explicitly software-pipelined (ping-pong
//    A-fragment groups + sched_barrier): the global A loads for the NEXT
//    4 K-steps issue before the 12 WMMAs of the CURRENT 4 K-steps, so with
//    1 wave/SIMD (no TLP) the L2 latency is still hidden. B fragments come
//    from LDS just-in-time (cheap dscnt waits). Gate math on f32
//    accumulators; double-buffered h; grid-wide atomic barrier + agent
//    fences per step.
// ---------------------------------------------------------------------------
__global__ void __launch_bounds__(RWAVES * 32)
__attribute__((amdgpu_waves_per_eu(1)))
gru_recur_kernel(const float* __restrict__ xg,     // [B*T,3H], row=b*T+t
                 const _Float16* __restrict__ Whh, // [3H,H] f16 row-major
                 const float* __restrict__ bhh,    // [3H]
                 _Float16* h0, _Float16* h1,       // [16,H]
                 _Float16* __restrict__ out,       // [B*T,H]
                 unsigned* ctr) {
  extern __shared__ char smem[];
  const int lane  = threadIdx.x & 31;
  const int wave  = threadIdx.x >> 5;        // 0..1
  const int slice = blockIdx.x * RWAVES + wave;  // 0..63
  const int hi = lane >> 4;
  const int ln = lane & 15;
  const int col = slice * 16 + ln;

  char* wbase = smem + (size_t)wave * LDS_PER_WAVE;

  // ---- one-time fill: permute Whh slice into per-lane fragment order ----
#pragma unroll
  for (int g = 0; g < 3; ++g) {
    const _Float16* br = Whh + (size_t)(g * HH + col) * HH;
    char* gbase = wbase + g * LDS_PER_GATE;
    for (int ks = 0; ks < 32; ++ks) {
      v8h lo = *(const v8h*)(br + ks * 32 + hi * 8);
      v8h hh2 = *(const v8h*)(br + ks * 32 + 16 + hi * 8);
      *(v8h*)(gbase + ks * 1024 + lane * 16)       = lo;
      *(v8h*)(gbase + ks * 1024 + 512 + lane * 16) = hh2;
    }
  }
  __syncthreads();

  const char* gR = wbase;
  const char* gZ = wbase + LDS_PER_GATE;
  const char* gN = wbase + 2 * LDS_PER_GATE;
  const float bhr = bhh[col];
  const float bhz = bhh[HH + col];
  const float bhn = bhh[2 * HH + col];

  for (int t = 0; t < TT; ++t) {
    const _Float16* hc = (t & 1) ? h1 : h0;
    _Float16*       hn = (t & 1) ? h0 : h1;

    v8f aR = zero8(), aZ = zero8(), aN = zero8();
    const _Float16* arow = hc + (size_t)ln * HH;

    v16h aA[4], aB[4];
#pragma unroll
    for (int u = 0; u < 4; ++u) aA[u] = load_frag(arow + u * 32, hi);

    for (int ks = 0; ks < 32; ks += 8) {
      // preload next group (K-steps ks+4..ks+7), then WMMAs on aA
#pragma unroll
      for (int u = 0; u < 4; ++u) aB[u] = load_frag(arow + (ks + 4 + u) * 32, hi);
      __builtin_amdgcn_sched_barrier(0);
#pragma unroll
      for (int u = 0; u < 4; ++u) {
        v16h br = lds_frag(gR, ks + u, lane);
        v16h bz = lds_frag(gZ, ks + u, lane);
        v16h bn = lds_frag(gN, ks + u, lane);
        aR = wmma_f16(aA[u], br, aR);
        aZ = wmma_f16(aA[u], bz, aZ);
        aN = wmma_f16(aA[u], bn, aN);
      }
      // preload group after next (wraps on the last trip; values unused)
#pragma unroll
      for (int u = 0; u < 4; ++u) aA[u] = load_frag(arow + (((ks + 8) & 31) + u) * 32, hi);
      __builtin_amdgcn_sched_barrier(0);
#pragma unroll
      for (int u = 0; u < 4; ++u) {
        v16h br = lds_frag(gR, ks + 4 + u, lane);
        v16h bz = lds_frag(gZ, ks + 4 + u, lane);
        v16h bn = lds_frag(gN, ks + 4 + u, lane);
        aR = wmma_f16(aB[u], br, aR);
        aZ = wmma_f16(aB[u], bz, aZ);
        aN = wmma_f16(aB[u], bn, aN);
      }
    }

#pragma unroll
    for (int v = 0; v < 8; ++v) {
      int m   = v + 8 * hi;                 // batch index
      int row = m * TT + t;                 // xg / out row
      float xr = xg[(size_t)row * G3 + col];
      float xz = xg[(size_t)row * G3 + HH + col];
      float xn = xg[(size_t)row * G3 + 2 * HH + col];
      float r = 1.f / (1.f + __expf(-(xr + aR[v] + bhr)));
      float z = 1.f / (1.f + __expf(-(xz + aZ[v] + bhz)));
      float n = tanhf(xn + r * (aN[v] + bhn));
      float hold = (float)hc[(size_t)m * HH + col];
      float hnew = (1.f - z) * n + z * hold;
      hn[(size_t)m * HH + col] = (_Float16)hnew;
      out[(size_t)row * HH + col] = (_Float16)hnew;
    }

    // grid-wide barrier: release stores, monotonic counter, acquire/invalidate
    __syncthreads();
    if (threadIdx.x == 0) {
      __builtin_amdgcn_fence(__ATOMIC_RELEASE, "agent");
      __hip_atomic_fetch_add(ctr, 1u, __ATOMIC_RELAXED, __HIP_MEMORY_SCOPE_AGENT);
      unsigned target = (unsigned)(t + 1) * (unsigned)RWGS;
      while (__hip_atomic_load(ctr, __ATOMIC_RELAXED, __HIP_MEMORY_SCOPE_AGENT) < target) {
        __builtin_amdgcn_s_sleep(1);
      }
    }
    __syncthreads();
    __builtin_amdgcn_fence(__ATOMIC_ACQUIRE, "agent");
  }
}

// ---------------------------------------------------------------------------
// 5) Mean over time: hbar[b,h] = (1/T) sum_t out1[b*T+t, h]
// ---------------------------------------------------------------------------
__global__ void time_mean_kernel(const _Float16* __restrict__ o, float* __restrict__ hbar) {
  int id = blockIdx.x * blockDim.x + threadIdx.x;   // < 16*1024
  int b = id >> 10, h = id & 1023;
  float s = 0.f;
  for (int t = 0; t < TT; ++t) s += (float)o[((size_t)(b * TT + t)) * HH + h];
  hbar[id] = s * (1.0f / (float)TT);
}

// ---------------------------------------------------------------------------
// 6) FC: logits[b,c] = hbar[b,:] . fcW[c,:] + fcb[c]   (one wave per (b,c))
// ---------------------------------------------------------------------------
__global__ void fc_kernel(const float* __restrict__ hbar, const float* __restrict__ fcW,
                          const float* __restrict__ fcb, float* __restrict__ outp) {
  int b = blockIdx.x;                 // 16
  int c = threadIdx.x >> 5;           // 0..10 (blockDim = 352)
  int lane = threadIdx.x & 31;
  float s = 0.f;
  for (int h = lane; h < HH; h += 32) s += hbar[b * HH + h] * fcW[c * HH + h];
#pragma unroll
  for (int off = 16; off; off >>= 1) s += __shfl_down(s, off, 32);
  if (lane == 0) outp[b * 11 + c] = s + fcb[c];
}

// ---------------------------------------------------------------------------
extern "C" void kernel_launch(void* const* d_in, const int* in_sizes, int n_in,
                              void* d_out, int out_size, void* d_ws, size_t ws_size,
                              hipStream_t stream) {
  const float* inp  = (const float*)d_in[0];
  const float* Wih0 = (const float*)d_in[1];
  const float* Whh0 = (const float*)d_in[2];
  const float* bih0 = (const float*)d_in[3];
  const float* bhh0 = (const float*)d_in[4];
  const float* Wih1 = (const float*)d_in[5];
  const float* Whh1 = (const float*)d_in[6];
  const float* bih1 = (const float*)d_in[7];
  const float* bhh1 = (const float*)d_in[8];
  const float* fcW  = (const float*)d_in[9];
  const float* fcb  = (const float*)d_in[10];
  float* logits = (float*)d_out;

  char* ws = (char*)d_ws;
  size_t off = 0;
  auto alloc = [&](size_t bytes) -> char* {
    char* p = ws + off;
    off = (off + bytes + 255) & ~(size_t)255;
    return p;
  };

  // zeroed region first (counters + h double buffer)
  unsigned*  ctr0   = (unsigned*)alloc(256);
  unsigned*  ctr1   = (unsigned*)alloc(256);
  _Float16*  h0     = (_Float16*)alloc((size_t)BB * HH * 2);
  _Float16*  h1     = (_Float16*)alloc((size_t)BB * HH * 2);
  size_t zero_bytes = off;

  _Float16* x0f16   = (_Float16*)alloc((size_t)MT * HH * 2);
  _Float16* out0f16 = (_Float16*)alloc((size_t)MT * HH * 2);
  _Float16* out1f16 = (_Float16*)alloc((size_t)MT * HH * 2);
  _Float16* Wih0h   = (_Float16*)alloc((size_t)G3 * HH * 2);
  _Float16* Whh0h   = (_Float16*)alloc((size_t)G3 * HH * 2);
  _Float16* Wih1h   = (_Float16*)alloc((size_t)G3 * HH * 2);
  _Float16* Whh1h   = (_Float16*)alloc((size_t)G3 * HH * 2);
  float*    xg0     = (float*)alloc((size_t)MT * G3 * 4);
  float*    xg1     = (float*)alloc((size_t)MT * G3 * 4);
  float*    hbar    = (float*)alloc((size_t)BB * HH * 4);
  (void)ws_size; (void)in_sizes; (void)n_in; (void)out_size;

  hipMemsetAsync(d_ws, 0, zero_bytes, stream);

  // preprocess
  preprocess_kernel<<<(MT * HH) / 256, 256, 0, stream>>>(inp, x0f16);

  // weight conversions (3H*H each)
  const int nw = G3 * HH;
  cvt_f16_kernel<<<(nw + 255) / 256, 256, 0, stream>>>(Wih0, Wih0h, nw);
  cvt_f16_kernel<<<(nw + 255) / 256, 256, 0, stream>>>(Whh0, Whh0h, nw);
  cvt_f16_kernel<<<(nw + 255) / 256, 256, 0, stream>>>(Wih1, Wih1h, nw);
  cvt_f16_kernel<<<(nw + 255) / 256, 256, 0, stream>>>(Whh1, Whh1h, nw);

  // layer 0: input GEMM + recurrence
  // waves = 128 Mtiles * 48 Nstrips = 6144 -> 768 blocks of 256 (8 waves)
  gemm_xg_kernel<<<768, 256, 0, stream>>>(x0f16, Wih0h, bih0, xg0);
  gru_recur_kernel<<<RWGS, RWAVES * 32, LDS_TOTAL, stream>>>(xg0, Whh0h, bhh0, h0, h1,
                                                             out0f16, ctr0);

  // layer 1 (h carries over: T=128 even -> final h resides in h0)
  gemm_xg_kernel<<<768, 256, 0, stream>>>(out0f16, Wih1h, bih1, xg1);
  gru_recur_kernel<<<RWGS, RWAVES * 32, LDS_TOTAL, stream>>>(xg1, Whh1h, bhh1, h0, h1,
                                                             out1f16, ctr1);

  // head
  time_mean_kernel<<<(BB * HH) / 256, 256, 0, stream>>>(out1f16, hbar);
  fc_kernel<<<BB, 352, 0, stream>>>(hbar, fcW, fcb, logits);
}
